// Model_lite_22033182228934
// MI455X (gfx1250) — matrix operand order repaired and checked by reading, not compiled, not run on hardware
//
#include <hip/hip_runtime.h>
#include <hip/hip_bf16.h>
#include <cstdint>

// ---------------------------------------------------------------------------
// MI455X (gfx1250) transformer layer:
//   * All four GEMMs via v_wmma_f32_16x16x32_bf16 (fp32 read -> bf16 LDS tiles)
//     with ping-pong LDS double buffering (global loads overlap WMMA compute)
//   * Flash-attention per (head, 64-row query tile), QK^T and P*V via WMMA
//   * wave32 layouts per CDNA5 ISA 7.12.2
// ---------------------------------------------------------------------------

constexpr int SEQ   = 2048;
constexpr int HID   = 4096;
constexpr int NHEAD = 32;
constexpr int HDIM  = 128;
constexpr int INNER = 16384;

// layer 0 constants
constexpr float ALPHA_C   = 7.4833147735478830f;   // sqrt(2*28)
constexpr float SCALING_C = 1.0f;                   // layer_id + 1
constexpr float INV_COEFF_SCALED = 0.0883883476483184f; // SCALING / (sqrt(128)*SCALING)
constexpr float MASKED_VAL = -100.0f * SCALING_C;
constexpr float LN_EPS = 1e-5f;

typedef __bf16 bfx16 __attribute__((ext_vector_type(16)));
typedef float  fx8   __attribute__((ext_vector_type(8)));

union FragU {
  bfx16 v;
  uint4 q[2];
  unsigned short u[16];
};

static __device__ __forceinline__ unsigned short f2bf(float f) {
  unsigned int u = __float_as_uint(f);
  unsigned int r = (u + 0x7fffu + ((u >> 16) & 1u)) >> 16;  // RNE
  return (unsigned short)r;
}

static __device__ __forceinline__ fx8 wmma_bf16(const FragU& a, const FragU& b, fx8 c) {
  return __builtin_amdgcn_wmma_f32_16x16x32_bf16(false, a.v, false, b.v,
                                                 (short)0, c, false, false);
}

static __device__ __forceinline__ float gelu_tanh(float x) {
  float x3 = x * x * x;
  return 0.5f * x * (1.0f + tanhf(0.7978845608028654f * (x + 0.044715f * x3)));
}

// ---------------------------------------------------------------------------
// LayerNorm: one row (4096) per 256-thread block
// ---------------------------------------------------------------------------
__global__ __launch_bounds__(256) void ln_kernel(const float* __restrict__ x,
                                                 const float* __restrict__ w,
                                                 const float* __restrict__ b,
                                                 float* __restrict__ y) {
  __shared__ float r1[256];
  __shared__ float r2[256];
  const int row = blockIdx.x;
  const int tid = threadIdx.x;
  const float* xr = x + (size_t)row * HID;
  float s = 0.f, s2 = 0.f;
  for (int i = tid; i < HID; i += 256) { float v = xr[i]; s += v; s2 += v * v; }
  r1[tid] = s; r2[tid] = s2;
  __syncthreads();
  for (int st = 128; st > 0; st >>= 1) {
    if (tid < st) { r1[tid] += r1[tid + st]; r2[tid] += r2[tid + st]; }
    __syncthreads();
  }
  const float mean = r1[0] * (1.0f / HID);
  const float var  = r2[0] * (1.0f / HID) - mean * mean;
  const float inv  = rsqrtf(var + LN_EPS);
  float* yr = y + (size_t)row * HID;
  for (int i = tid; i < HID; i += 256)
    yr[i] = (xr[i] - mean) * inv * w[i] + b[i];
}

// hidden = residual*ALPHA + attn_out, then LayerNorm -> y
__global__ __launch_bounds__(256) void add_ln_kernel(const float* __restrict__ res,
                                                     const float* __restrict__ att,
                                                     const float* __restrict__ w,
                                                     const float* __restrict__ b,
                                                     float* __restrict__ y) {
  __shared__ float r1[256];
  __shared__ float r2[256];
  const int row = blockIdx.x;
  const int tid = threadIdx.x;
  const float* ar = res + (size_t)row * HID;
  const float* br = att + (size_t)row * HID;
  float s = 0.f, s2 = 0.f;
  for (int i = tid; i < HID; i += 256) {
    float v = ar[i] * ALPHA_C + br[i]; s += v; s2 += v * v;
  }
  r1[tid] = s; r2[tid] = s2;
  __syncthreads();
  for (int st = 128; st > 0; st >>= 1) {
    if (tid < st) { r1[tid] += r1[tid + st]; r2[tid] += r2[tid + st]; }
    __syncthreads();
  }
  const float mean = r1[0] * (1.0f / HID);
  const float var  = r2[0] * (1.0f / HID) - mean * mean;
  const float inv  = rsqrtf(var + LN_EPS);
  float* yr = y + (size_t)row * HID;
  for (int i = tid; i < HID; i += 256)
    yr[i] = (ar[i] * ALPHA_C + br[i] - mean) * inv * w[i] + b[i];
}

// out = res2*ALPHA + mlp_out
__global__ void final_add_kernel(const float* __restrict__ res2,
                                 const float* __restrict__ mlp,
                                 float* __restrict__ out) {
  size_t i = (size_t)blockIdx.x * blockDim.x + threadIdx.x;
  out[i] = res2[i] * ALPHA_C + mlp[i];
}

// ---------------------------------------------------------------------------
// GEMM: C[M,N] = A[M,K] * W[N,K]^T + bias  (both operands K-contiguous)
// 256 threads = 8 waves; block tile 128x128x32; wave tile 64x32.
// fp32 global -> bf16 LDS (ping-pong double buffer) -> WMMA f32 accumulate.
// Global loads for tile k+1 are issued before the WMMA burst on tile k, so
// HBM latency hides behind matrix ops (workload is near the 23.3 TB/s roof).
// EPI: 0 = bias, 1 = bias + gelu(tanh)
// ---------------------------------------------------------------------------
template <int EPI>
__global__ __launch_bounds__(256) void gemm_bf16_kernel(const float* __restrict__ A,
                                                        const float* __restrict__ W,
                                                        const float* __restrict__ bias,
                                                        float* __restrict__ C,
                                                        int M, int N, int K) {
  constexpr int BM = 128, BN = 128, BK = 32, PITCH = BK + 8;  // 40 shorts = 80B rows
  __shared__ alignas(16) unsigned short As[2][BM][PITCH];
  __shared__ alignas(16) unsigned short Bs[2][BN][PITCH];

  const int m0 = blockIdx.x * BM;   // x-fast: co-resident blocks share W strip (L2 reuse)
  const int n0 = blockIdx.y * BN;
  const int tid  = threadIdx.x;
  const int wid  = tid >> 5;
  const int lane = tid & 31;
  const int lr   = lane & 15;
  const int hi   = lane >> 4;
  const int wm = (wid >> 2) * 64;   // wave m offset within block
  const int wn = (wid & 3) * 32;    // wave n offset within block

  // cooperative-load coordinates (4 float4 per thread per matrix per tile)
  int ldr[4], ldk[4];
  #pragma unroll
  for (int i = 0; i < 4; ++i) {
    const int fid = tid + i * 256;
    ldr[i] = fid >> 3;
    ldk[i] = (fid & 7) << 2;
  }

  float4 sa[4], sw[4];  // register staging for the in-flight tile

  auto issue_loads = [&](int k0) {
    #pragma unroll
    for (int i = 0; i < 4; ++i) {
      sa[i] = *reinterpret_cast<const float4*>(&A[(size_t)(m0 + ldr[i]) * K + k0 + ldk[i]]);
      sw[i] = *reinterpret_cast<const float4*>(&W[(size_t)(n0 + ldr[i]) * K + k0 + ldk[i]]);
    }
  };
  auto store_tile = [&](int buf) {
    #pragma unroll
    for (int i = 0; i < 4; ++i) {
      ushort4 ap; ap.x = f2bf(sa[i].x); ap.y = f2bf(sa[i].y);
      ap.z = f2bf(sa[i].z); ap.w = f2bf(sa[i].w);
      *reinterpret_cast<ushort4*>(&As[buf][ldr[i]][ldk[i]]) = ap;
      ushort4 wp; wp.x = f2bf(sw[i].x); wp.y = f2bf(sw[i].y);
      wp.z = f2bf(sw[i].z); wp.w = f2bf(sw[i].w);
      *reinterpret_cast<ushort4*>(&Bs[buf][ldr[i]][ldk[i]]) = wp;
    }
  };

  fx8 acc[4][2];
  #pragma unroll
  for (int mt = 0; mt < 4; ++mt)
    #pragma unroll
    for (int nt = 0; nt < 2; ++nt)
      acc[mt][nt] = (fx8){0.f, 0.f, 0.f, 0.f, 0.f, 0.f, 0.f, 0.f};

  // prologue: stage + store tile 0
  issue_loads(0);
  store_tile(0);
  int buf = 0;

  for (int k0 = 0; k0 < K; k0 += BK) {
    __syncthreads();                 // tile `buf` visible; buf^1 free to overwrite
    const bool more = (k0 + BK) < K;
    if (more) issue_loads(k0 + BK);  // loads in flight during the WMMA burst

    // A fragments: row = lane&15, K halves split by lane half (ISA 16-bit A layout)
    FragU af[4];
    #pragma unroll
    for (int mt = 0; mt < 4; ++mt) {
      const unsigned short* ap = &As[buf][wm + mt * 16 + lr][hi * 8];
      af[mt].q[0] = *reinterpret_cast<const uint4*>(ap);        // K = base..base+7
      af[mt].q[1] = *reinterpret_cast<const uint4*>(ap + 16);   // K = base+16..base+23
    }
    // B fragments: col = lane&15, K = hi*8 + e (e<8), 16 + hi*8 + (e-8) (e>=8)
    FragU bf[2];
    #pragma unroll
    for (int nt = 0; nt < 2; ++nt) {
      const unsigned short* bp = &Bs[buf][wn + nt * 16 + lr][hi * 8];
      bf[nt].q[0] = *reinterpret_cast<const uint4*>(bp);
      bf[nt].q[1] = *reinterpret_cast<const uint4*>(bp + 16);
    }
    #pragma unroll
    for (int mt = 0; mt < 4; ++mt)
      #pragma unroll
      for (int nt = 0; nt < 2; ++nt)
        acc[mt][nt] = wmma_bf16(af[mt], bf[nt], acc[mt][nt]);

    if (more) store_tile(buf ^ 1);   // convert staged regs into the other buffer
    buf ^= 1;
  }

  // epilogue: C layout => N = lane&15, M = r + 8*hi
  #pragma unroll
  for (int mt = 0; mt < 4; ++mt) {
    #pragma unroll
    for (int nt = 0; nt < 2; ++nt) {
      const int n = n0 + wn + nt * 16 + lr;
      const float bv = bias[n];
      #pragma unroll
      for (int r = 0; r < 8; ++r) {
        const int m = m0 + wm + mt * 16 + hi * 8 + r;
        float v = acc[mt][nt][r] + bv;
        if (EPI == 1) v = gelu_tanh(v);
        C[(size_t)m * N + n] = v;
      }
    }
  }
}

// ---------------------------------------------------------------------------
// RoPE split: qkv[S, NH, 384] -> q/k/v buffers laid out [head][token][128]
// ---------------------------------------------------------------------------
__global__ __launch_bounds__(128) void rope_kernel(const float* __restrict__ qkv,
                                                   const int* __restrict__ pos_ids,
                                                   const int* __restrict__ blk_ids,
                                                   const float* __restrict__ cosc,
                                                   const float* __restrict__ sinc,
                                                   float* __restrict__ qb,
                                                   float* __restrict__ kb,
                                                   float* __restrict__ vb) {
  const int t = blockIdx.x;
  const int h = blockIdx.y;
  const int d = threadIdx.x;  // 0..127
  const float* base = qkv + (size_t)t * (NHEAD * 384) + (size_t)h * 384;
  const float qv = base[d];
  const float kv = base[128 + d];
  const float vv = base[256 + d];

  const int half = (d >= 64);       // 0: positional, 1: block tables
  const int dd   = d - half * 64;   // 0..63 within its 64-wide rope group
  const int pid  = half ? blk_ids[t] : pos_ids[t];
  const float c  = cosc[(size_t)pid * 64 + dd];
  const float s  = sinc[(size_t)pid * 64 + dd];
  // rotate_half within the 64-wide group
  const int off  = half * 64;
  const float q_rot = (dd < 32) ? -base[off + dd + 32]       : base[off + dd - 32];
  const float k_rot = (dd < 32) ? -base[128 + off + dd + 32] : base[128 + off + dd - 32];

  const size_t oidx = ((size_t)h * SEQ + t) * HDIM + d;
  qb[oidx] = qv * c + q_rot * s;
  kb[oidx] = kv * c + k_rot * s;
  vb[oidx] = vv;
}

// ---------------------------------------------------------------------------
// Flash attention: block = (head, 64-query tile); 4 waves, 16 rows each.
// QK^T and P*V via bf16 WMMA; online softmax with shfl_xor row reductions.
// ---------------------------------------------------------------------------
__global__ __launch_bounds__(128) void attn_kernel(const float* __restrict__ qb,
                                                   const float* __restrict__ kb,
                                                   const float* __restrict__ vb,
                                                   const unsigned char* __restrict__ mask,
                                                   float* __restrict__ ctx) {
  __shared__ alignas(16) unsigned short Ks[64][136];   // [key][d]   272B pitch
  __shared__ alignas(16) unsigned short Vt[128][72];   // [d][key]   144B pitch
  __shared__ alignas(16) unsigned short Ps[64][72];    // [qrow][key]144B pitch

  const int qt = blockIdx.x, h = blockIdx.y;
  const int tid  = threadIdx.x;
  const int wid  = tid >> 5;
  const int lane = tid & 31;
  const int lr   = lane & 15;
  const int hi   = lane >> 4;
  const int q0   = qt * 64;

  const float* Q  = qb + ((size_t)h * SEQ + q0) * HDIM;
  const float* Kg = kb + (size_t)h * SEQ * HDIM;
  const float* Vg = vb + (size_t)h * SEQ * HDIM;

  // Q fragments (A layout), one 16x128 strip per wave, 4 K-steps of 32
  FragU qf[4];
  {
    const int qrow = wid * 16 + lr;
    #pragma unroll
    for (int kk = 0; kk < 4; ++kk) {
      const float* qp = Q + (size_t)qrow * HDIM + kk * 32 + hi * 8;
      #pragma unroll
      for (int e = 0; e < 8; ++e) qf[kk].u[e] = f2bf(qp[e]);
      #pragma unroll
      for (int e = 0; e < 8; ++e) qf[kk].u[8 + e] = f2bf(qp[16 + e]);
    }
  }

  fx8 o[8];
  #pragma unroll
  for (int i = 0; i < 8; ++i) o[i] = (fx8){0.f, 0.f, 0.f, 0.f, 0.f, 0.f, 0.f, 0.f};
  float mrun[8], lrun[8];
  #pragma unroll
  for (int r = 0; r < 8; ++r) { mrun[r] = -1e30f; lrun[r] = 0.f; }

  for (int j = 0; j < SEQ / 64; ++j) {
    __syncthreads();
    // cooperative load K (row-major) and V (transposed) into LDS as bf16
    #pragma unroll
    for (int i = 0; i < 16; ++i) {
      const int fid = tid + i * 128;       // 0..2047 float4 ids
      const int r   = fid >> 5;
      const int c4  = (fid & 31) << 2;
      const float4 k4 = *reinterpret_cast<const float4*>(Kg + ((size_t)(j * 64 + r)) * HDIM + c4);
      ushort4 kp; kp.x = f2bf(k4.x); kp.y = f2bf(k4.y); kp.z = f2bf(k4.z); kp.w = f2bf(k4.w);
      *reinterpret_cast<ushort4*>(&Ks[r][c4]) = kp;
      const float4 v4 = *reinterpret_cast<const float4*>(Vg + ((size_t)(j * 64 + r)) * HDIM + c4);
      Vt[c4 + 0][r] = f2bf(v4.x);
      Vt[c4 + 1][r] = f2bf(v4.y);
      Vt[c4 + 2][r] = f2bf(v4.z);
      Vt[c4 + 3][r] = f2bf(v4.w);
    }
    // prefetch next K/V tile toward L2 (global_prefetch_b8)
    if (j + 1 < SEQ / 64) {
      const size_t nbase = ((size_t)(j + 1) * 64 + (tid >> 1)) * HDIM + (tid & 1) * 64;
      __builtin_prefetch(Kg + nbase, 0, 1);
      __builtin_prefetch(Vg + nbase, 0, 1);
    }
    __syncthreads();

    // S = Q * K^T : 4 column tiles of 16 keys, 4 K-steps of 32 over d
    fx8 sa[4];
    #pragma unroll
    for (int ct = 0; ct < 4; ++ct) {
      sa[ct] = (fx8){0.f, 0.f, 0.f, 0.f, 0.f, 0.f, 0.f, 0.f};
      #pragma unroll
      for (int kk = 0; kk < 4; ++kk) {
        FragU bfr;
        const unsigned short* bp = &Ks[ct * 16 + lr][kk * 32 + hi * 8];
        bfr.q[0] = *reinterpret_cast<const uint4*>(bp);
        bfr.q[1] = *reinterpret_cast<const uint4*>(bp + 16);
        sa[ct] = wmma_bf16(qf[kk], bfr, sa[ct]);
      }
    }

    // mask + scale (scores/COEFF, mask -> -100, then *SCALING folded)
    #pragma unroll
    for (int ct = 0; ct < 4; ++ct) {
      const int col = j * 64 + ct * 16 + lr;
      #pragma unroll
      for (int r = 0; r < 8; ++r) {
        const int qr = q0 + wid * 16 + hi * 8 + r;
        float s = sa[ct][r] * INV_COEFF_SCALED;
        if (mask[(size_t)qr * SEQ + col]) s = MASKED_VAL;
        sa[ct][r] = s;
      }
    }

    // online softmax: row stats reduced across the 16 lanes of each half
    float alpha[8];
    #pragma unroll
    for (int r = 0; r < 8; ++r) {
      float v = fmaxf(fmaxf(sa[0][r], sa[1][r]), fmaxf(sa[2][r], sa[3][r]));
      #pragma unroll
      for (int off = 1; off < 16; off <<= 1) v = fmaxf(v, __shfl_xor(v, off, 32));
      const float mn = fmaxf(mrun[r], v);
      alpha[r] = __expf(mrun[r] - mn);
      mrun[r]  = mn;
    }
    #pragma unroll
    for (int ct = 0; ct < 4; ++ct)
      #pragma unroll
      for (int r = 0; r < 8; ++r)
        sa[ct][r] = __expf(sa[ct][r] - mrun[r]);
    #pragma unroll
    for (int r = 0; r < 8; ++r) {
      float v = sa[0][r] + sa[1][r] + sa[2][r] + sa[3][r];
      #pragma unroll
      for (int off = 1; off < 16; off <<= 1) v += __shfl_xor(v, off, 32);
      lrun[r] = lrun[r] * alpha[r] + v;
    }
    #pragma unroll
    for (int nt = 0; nt < 8; ++nt)
      #pragma unroll
      for (int r = 0; r < 8; ++r)
        o[nt][r] *= alpha[r];

    // stage P (wave-private LDS region) then O += P * V
    #pragma unroll
    for (int ct = 0; ct < 4; ++ct)
      #pragma unroll
      for (int r = 0; r < 8; ++r)
        Ps[wid * 16 + hi * 8 + r][ct * 16 + lr] = f2bf(sa[ct][r]);

    #pragma unroll
    for (int kk = 0; kk < 2; ++kk) {
      FragU pf;
      const unsigned short* pp = &Ps[wid * 16 + lr][kk * 32 + hi * 8];
      pf.q[0] = *reinterpret_cast<const uint4*>(pp);
      pf.q[1] = *reinterpret_cast<const uint4*>(pp + 16);
      #pragma unroll
      for (int nt = 0; nt < 8; ++nt) {
        FragU vf;
        const unsigned short* vp = &Vt[nt * 16 + lr][kk * 32 + hi * 8];
        vf.q[0] = *reinterpret_cast<const uint4*>(vp);
        vf.q[1] = *reinterpret_cast<const uint4*>(vp + 16);
        o[nt] = wmma_bf16(pf, vf, o[nt]);
      }
    }
  }

  // context[q, h*128 + d] = O / l
  #pragma unroll
  for (int nt = 0; nt < 8; ++nt) {
    const int d = nt * 16 + lr;
    #pragma unroll
    for (int r = 0; r < 8; ++r) {
      const int qr = q0 + wid * 16 + hi * 8 + r;
      ctx[(size_t)qr * HID + h * HDIM + d] = o[nt][r] / lrun[r];
    }
  }
}

// ---------------------------------------------------------------------------
extern "C" void kernel_launch(void* const* d_in, const int* in_sizes, int n_in,
                              void* d_out, int out_size, void* d_ws, size_t ws_size,
                              hipStream_t stream) {
  (void)in_sizes; (void)n_in; (void)out_size; (void)ws_size;
  const float* hidden   = (const float*)d_in[0];
  const int*   pos_ids  = (const int*)d_in[1];
  const int*   blk_ids  = (const int*)d_in[2];
  const unsigned char* mask = (const unsigned char*)d_in[3];
  const float* cosc  = (const float*)d_in[4];
  const float* sinc  = (const float*)d_in[5];
  const float* ln1w  = (const float*)d_in[6];
  const float* ln1b  = (const float*)d_in[7];
  const float* qkvw  = (const float*)d_in[8];
  const float* qkvb  = (const float*)d_in[9];
  const float* densew = (const float*)d_in[10];
  const float* denseb = (const float*)d_in[11];
  const float* ln2w  = (const float*)d_in[12];
  const float* ln2b  = (const float*)d_in[13];
  const float* w1    = (const float*)d_in[14];
  const float* b1    = (const float*)d_in[15];
  const float* w2    = (const float*)d_in[16];
  const float* b2    = (const float*)d_in[17];
  float* out = (float*)d_out;

  // workspace layout (floats)
  const size_t SH = (size_t)SEQ * HID;          // 8M
  float* ws       = (float*)d_ws;
  float* residual = ws;                          // LN1 output
  float* qkvbuf   = residual + SH;               // 24M
  float* q_buf    = qkvbuf + (size_t)SEQ * 3 * HID;
  float* k_buf    = q_buf + SH;                  // [NH][S][128] each
  float* v_buf    = k_buf + SH;
  float* context  = v_buf + SH;
  float* attn_out = context + SH;
  float* res2     = attn_out + SH;
  float* h_buf    = res2 + SH;                   // 32M
  float* mlp_out  = h_buf + (size_t)SEQ * INNER;

  // 1) LN1
  ln_kernel<<<SEQ, 256, 0, stream>>>(hidden, ln1w, ln1b, residual);
  // 2) QKV GEMM (M=2048, N=12288, K=4096)
  gemm_bf16_kernel<0><<<dim3(SEQ / 128, (3 * HID) / 128), 256, 0, stream>>>(
      residual, qkvw, qkvb, qkvbuf, SEQ, 3 * HID, HID);
  // 3) RoPE split into head-major q/k/v
  rope_kernel<<<dim3(SEQ, NHEAD), 128, 0, stream>>>(qkvbuf, pos_ids, blk_ids,
                                                    cosc, sinc, q_buf, k_buf, v_buf);
  // 4) flash attention
  attn_kernel<<<dim3(SEQ / 64, NHEAD), 128, 0, stream>>>(q_buf, k_buf, v_buf, mask, context);
  // 5) dense projection
  gemm_bf16_kernel<0><<<dim3(SEQ / 128, HID / 128), 256, 0, stream>>>(
      context, densew, denseb, attn_out, SEQ, HID, HID);
  // 6) residual*ALPHA + attn, LN2
  add_ln_kernel<<<SEQ, 256, 0, stream>>>(residual, attn_out, ln2w, ln2b, res2);
  // 7) MLP up + GELU fused
  gemm_bf16_kernel<1><<<dim3(SEQ / 128, INNER / 128), 256, 0, stream>>>(
      res2, w1, b1, h_buf, SEQ, INNER, HID);
  // 8) MLP down
  gemm_bf16_kernel<0><<<dim3(SEQ / 128, HID / 128), 256, 0, stream>>>(
      h_buf, w2, b2, mlp_out, SEQ, HID, INNER);
  // 9) final residual
  final_add_kernel<<<(int)(SH / 256), 256, 0, stream>>>(res2, mlp_out, out);
}
